// BNBEmbeddingWithAdapter_90443421319841
// MI455X (gfx1250) — compile-verified, matrix operand order checked by reading, not run on hardware
//
#include <hip/hip_runtime.h>
#include <hip/hip_bf16.h>

// CDNA5 / gfx1250: wave32, WMMA 16x16x32 f16 -> f32
typedef _Float16 v16h __attribute__((ext_vector_type(16)));
typedef float    v8f  __attribute__((ext_vector_type(8)));

#define V_SZ   50400
#define D_SZ   4096
#define A_SZ   64
#define NTOK   8192         // B*S = 4*2048
#define TOK_TILE 16
#define COLS_PER_BLOCK 512
#define WSTRIDE 68          // 64 halves + 4 pad halves -> conflict-free fragment gathers

__global__ __launch_bounds__(256)
void bnb_emb_adapter_kernel(const int*   __restrict__ input_ids,
                            const int*   __restrict__ weight_q,     // uint8 codes stored as int32
                            const float* __restrict__ absmax,
                            const float* __restrict__ code,
                            const float* __restrict__ adapter_emb,  // [V, 64]
                            const float* __restrict__ adapter_W,    // [4096, 64]
                            float*       __restrict__ out)          // [8192, 4096]
{
    __shared__ float    s_code[256];
    __shared__ int      s_tok[TOK_TILE];
    __shared__ float    s_am[TOK_TILE];
    __shared__ _Float16 s_A[TOK_TILE * A_SZ];            // emb tile, [m][k] f16
    __shared__ _Float16 s_W[COLS_PER_BLOCK * WSTRIDE];   // W slice, [n][k] f16 (padded)

    const int tid      = threadIdx.x;
    const int lane     = tid & 31;          // wave32
    const int wave     = tid >> 5;          // 8 waves
    const int colGroup = blockIdx.x;        // 0..7   (512 columns each)
    const int tokTile  = blockIdx.y;        // 0..511 (16 tokens each)
    const int d0blk    = colGroup * COLS_PER_BLOCK;

    // ---- phase 1: code LUT, token ids + per-row absmax (BLOCK==D), W slice -> LDS f16
    s_code[tid] = code[tid];
    if (tid < TOK_TILE) {
        int t = input_ids[tokTile * TOK_TILE + tid];
        s_tok[tid] = t;
        s_am[tid]  = absmax[t];             // block index (t*D+d)/4096 == t since D==BLOCK
    }
    {
        const float4* W4 = (const float4*)adapter_W;    // rows are 16 float4
        #pragma unroll
        for (int i = 0; i < 32; ++i) {                  // 8192 float4 / 256 threads
            int idx = tid + i * 256;
            int n   = idx >> 4;                         // row within the 512-row slice
            int c4  = idx & 15;
            float4 v = W4[(size_t)(d0blk + n) * (A_SZ / 4) + c4];
            _Float16* dst = &s_W[n * WSTRIDE + c4 * 4];
            dst[0] = (_Float16)v.x; dst[1] = (_Float16)v.y;
            dst[2] = (_Float16)v.z; dst[3] = (_Float16)v.w;
        }
    }
    __syncthreads();

    // ---- phase 2: gather adapter_emb rows -> LDS f16; prefetch weight_q gather footprint
    {
        #pragma unroll
        for (int i = 0; i < 4; ++i) {                   // 1024 elems / 256 threads
            int idx = tid + i * 256;
            int m = idx >> 6, k = idx & 63;
            s_A[m * A_SZ + k] = (_Float16)adapter_emb[(size_t)s_tok[m] * A_SZ + k];
        }
        int m   = tid >> 4;                             // 16 rows x 16 segments of 128B
        int seg = tid & 15;
        __builtin_prefetch(weight_q + (size_t)s_tok[m] * D_SZ + d0blk + seg * 32, 0, 3);
    }
    __syncthreads();

    // ---- A fragments (16x32 f16 layout, ISA 7.12.2): shared by all 4 N-tiles of this wave
    const int mrow = lane & 15;
    const int kHi  = (lane & 16) >> 1;                  // upper half-wave holds K+8
    v16h a0, a1;
    #pragma unroll
    for (int h = 0; h < 16; ++h) {
        int k = ((h & 8) << 1) + (h & 7) + kHi;         // {0..7,16..23} (+8)
        a0[h] = s_A[mrow * A_SZ + k];
        a1[h] = s_A[mrow * A_SZ + 32 + k];
    }

    // per-lane epilogue rows: C/D layout puts M = r + 8*(lane>=16)
    const int mbase = (lane >> 4) << 3;
    int   tokq[8];
    float amq[8];
    #pragma unroll
    for (int r = 0; r < 8; ++r) {
        tokq[r] = s_tok[mbase + r];
        amq[r]  = s_am[mbase + r];
    }

    #pragma unroll
    for (int j = 0; j < 4; ++j) {                       // 4 N-tiles of 16 columns per wave
        const int d0   = d0blk + wave * 64 + j * 16;
        const int nloc = wave * 64 + j * 16 + mrow;     // B column N = lane&15
        v16h b0, b1;
        #pragma unroll
        for (int h = 0; h < 16; ++h) {
            int k = ((h & 8) << 1) + (h & 7) + kHi;
            b0[h] = s_W[nloc * WSTRIDE + k];
            b1[h] = s_W[nloc * WSTRIDE + 32 + k];
        }
        v8f c = {};
        c = __builtin_amdgcn_wmma_f32_16x16x32_f16(false, a0, false, b0, (short)0, c, false, false);
        c = __builtin_amdgcn_wmma_f32_16x16x32_f16(false, a1, false, b1, (short)0, c, false, false);

        const int col = d0 + mrow;
        #pragma unroll
        for (int r = 0; r < 8; ++r) {
            int   q   = weight_q[(size_t)tokq[r] * D_SZ + col] & 255;
            float val = c[r] + s_code[q] * amq[r];
            out[(size_t)(tokTile * TOK_TILE + mbase + r) * D_SZ + col] = val;
        }
    }
}

extern "C" void kernel_launch(void* const* d_in, const int* in_sizes, int n_in,
                              void* d_out, int out_size, void* d_ws, size_t ws_size,
                              hipStream_t stream) {
    const int*   input_ids   = (const int*)d_in[0];
    const int*   weight_q    = (const int*)d_in[1];
    const float* absmax      = (const float*)d_in[2];
    const float* code        = (const float*)d_in[3];
    const float* adapter_emb = (const float*)d_in[4];
    const float* adapter_W   = (const float*)d_in[5];
    float*       out         = (float*)d_out;

    dim3 grid(D_SZ / COLS_PER_BLOCK, NTOK / TOK_TILE);  // (8, 512)
    bnb_emb_adapter_kernel<<<grid, 256, 0, stream>>>(input_ids, weight_q, absmax, code,
                                                     adapter_emb, adapter_W, out);
}
